// Mamba2Block_12893491823336
// MI455X (gfx1250) — matrix operand order repaired and checked by reading, not compiled, not run on hardware
//
#include <hip/hip_runtime.h>

// ---------------------------------------------------------------------------
// Mamba block forward for MI455X (gfx1250, wave32, WMMA).
// GEMMs: register-tiled f16 WMMA (16x16x32, f32 accum). Each wave owns a
// 64x32 macro-tile (4x2 WMMA grid) -> 8 WMMAs per K-step on 6 fragment
// loads. N and K are template constants so all fragment/epilogue addresses
// fold into 24-bit instruction offsets; RESID is a template flag (no
// branches in the epilogue). __launch_bounds__(128,1) keeps accumulators
// in registers. Scan: one wave per (batch, channel), 2 SSM states per lane,
// shfl_xor reduction.
// ---------------------------------------------------------------------------

#define D_MODEL 1024
#define D_STATE 64
#define D_CONV  4
#define D_INNER 2048
#define DT_RANK 64
#define BATCH   2
#define SEQ_T   1024
#define MROWS   (BATCH * SEQ_T)         // 2048 token rows
#define XSSM_W  (DT_RANK + 2 * D_STATE) // 192

typedef __attribute__((ext_vector_type(16))) _Float16 v16h;
typedef __attribute__((ext_vector_type(8)))  _Float16 v8h;
typedef __attribute__((ext_vector_type(8)))  float    v8f;

// ---------------------------------------------------------------------------
// f32 -> f16 conversion (weights / activations)
// ---------------------------------------------------------------------------
__global__ void cvt_f32_f16(const float* __restrict__ in, _Float16* __restrict__ out, int n) {
    int i = blockIdx.x * blockDim.x + threadIdx.x;
    if (i < n) out[i] = (_Float16)in[i];
}

// ---------------------------------------------------------------------------
// LayerNorm over D_MODEL, emit f16 for the in_proj GEMM. One block per row.
// ---------------------------------------------------------------------------
__global__ void ln_f16_kernel(const float* __restrict__ x, const float* __restrict__ g,
                              const float* __restrict__ bb, _Float16* __restrict__ out) {
    const int D = D_MODEL;
    int row = blockIdx.x;
    int t = threadIdx.x;
    const float* xr = x + (size_t)row * D;
    float v[4];
    float s = 0.f, ss = 0.f;
#pragma unroll
    for (int j = 0; j < 4; ++j) {
        float f = xr[t + j * 256];
        v[j] = f; s += f; ss += f * f;
    }
#pragma unroll
    for (int m = 16; m >= 1; m >>= 1) {
        s  += __shfl_xor(s,  m, 32);
        ss += __shfl_xor(ss, m, 32);
    }
    __shared__ float sred[8], ssred[8], stats[2];
    int wid = t >> 5;
    if ((t & 31) == 0) { sred[wid] = s; ssred[wid] = ss; }
    __syncthreads();
    if (t == 0) {
        float S = 0.f, SS = 0.f;
        for (int i = 0; i < 8; ++i) { S += sred[i]; SS += ssred[i]; }
        float mu = S / D;
        float var = SS / D - mu * mu;
        stats[0] = mu;
        stats[1] = rsqrtf(var + 1e-5f);
    }
    __syncthreads();
    float mu = stats[0], rstd = stats[1];
    _Float16* orow = out + (size_t)row * D;
#pragma unroll
    for (int j = 0; j < 4; ++j) {
        int c = t + j * 256;
        orow[c] = (_Float16)(((v[j] - mu) * rstd) * g[c] + bb[c]);
    }
}

// ---------------------------------------------------------------------------
// Register-tiled WMMA GEMM: C[M,N] = A[M,K] * W[N,K]^T (+ resid[M,N]).
// One (16*TI)x(16*TJ) macro-tile per wave; blockDim.x = 128 (4 waves).
// MC/NC/KC compile-time: fragment + epilogue addresses are base + immediates.
// Fragments (16-bit A 16x32, B 32x16) use contiguous k = 16*half + s, same for A and B.
// ---------------------------------------------------------------------------
template <int TI, int TJ, int MC, int NC, int KC, bool RESID>
__global__ __launch_bounds__(128, 1)
void gemm_f16_wmma(const _Float16* __restrict__ A, const _Float16* __restrict__ W,
                   const float* __restrict__ resid, float* __restrict__ C) {
    constexpr int tilesN = NC / (16 * TJ);
    constexpr int totalTiles = (MC / (16 * TI)) * tilesN;
    int wid = blockIdx.x * 4 + (threadIdx.x >> 5);
    if (wid >= totalTiles) return;
    int tm = (wid / tilesN) * (16 * TI);
    int tn = (wid % tilesN) * (16 * TJ);
    int lane = threadIdx.x & 31;
    int half = lane >> 4;    // 0: lanes 0-15, 1: lanes 16-31
    int l15  = lane & 15;

    // Single per-operand base pointer; all row offsets are compile-time.
    const _Float16* Abase = A + (size_t)(tm + l15) * KC + half * 16;
    const _Float16* Wbase = W + (size_t)(tn + l15) * KC + half * 16;

    v8f acc[TI][TJ];
#pragma unroll
    for (int i = 0; i < TI; ++i)
#pragma unroll
        for (int j = 0; j < TJ; ++j) acc[i][j] = (v8f){};

    for (int k0 = 0; k0 < KC; k0 += 32) {
        v16h a[TI], b[TJ];
#pragma unroll
        for (int i = 0; i < TI; ++i) {
            // A 16x32 f16: half 0 -> K 0-15, half 1 -> K 16-31
            const _Float16* p = Abase + i * 16 * KC + k0;
            v8h lo = *(const v8h*)(p);
            v8h hi = *(const v8h*)(p + 8);
#pragma unroll
            for (int e = 0; e < 8; ++e) { a[i][e] = lo[e]; a[i][e + 8] = hi[e]; }
            __builtin_prefetch(p + 64, 0, 1);   // global_prefetch_b8, const offset
        }
#pragma unroll
        for (int j = 0; j < TJ; ++j) {
            // B 32x16 f16: half 0 -> K 0-15, half 1 -> K 16-31, N = l15
            const _Float16* p = Wbase + j * 16 * KC + k0;
            b[j] = *(const v16h*)(p);
            __builtin_prefetch(p + 64, 0, 1);
        }
#pragma unroll
        for (int i = 0; i < TI; ++i)
#pragma unroll
            for (int j = 0; j < TJ; ++j)
                acc[i][j] = __builtin_amdgcn_wmma_f32_16x16x32_f16(
                    false, a[i], false, b[j], (short)0, acc[i][j], false, false);
    }

    // Epilogue: one base per (i,j); r*NC folds into the 24-bit IOFFSET.
#pragma unroll
    for (int i = 0; i < TI; ++i) {
        // C layout: VGPR r -> M = tm+16i+r (lanes<16) / +8 (lanes>=16)
        size_t rowbase = (size_t)(tm + 16 * i + half * 8) * NC;
#pragma unroll
        for (int j = 0; j < TJ; ++j) {
            size_t base = rowbase + tn + 16 * j + l15;
            float* cp = C + base;
            const float* rp = RESID ? (resid + base) : nullptr;
#pragma unroll
            for (int r = 0; r < 8; ++r) {
                float v = acc[i][j][r];
                if (RESID) v += rp[r * NC];
                cp[r * NC] = v;
            }
        }
    }
}

// ---------------------------------------------------------------------------
// Causal depthwise conv (width 4) + SiLU. Emits f32 (scan) and f16 (x_proj).
// xz layout: [MROWS][2*D_INNER], xp = cols [0, D_INNER).
// ---------------------------------------------------------------------------
__global__ void conv_silu_kernel(const float* __restrict__ xz, const float* __restrict__ cw,
                                 const float* __restrict__ cb, _Float16* __restrict__ xch,
                                 float* __restrict__ xcf) {
    size_t i = (size_t)blockIdx.x * blockDim.x + threadIdx.x;
    if (i >= (size_t)MROWS * D_INNER) return;
    int d = (int)(i % D_INNER);
    size_t bt = i / D_INNER;
    int t = (int)(bt % SEQ_T);
    int b = (int)(bt / SEQ_T);
    float acc = cb[d];
#pragma unroll
    for (int k = 0; k < D_CONV; ++k) {
        int tt = t + k - (D_CONV - 1);
        if (tt >= 0)
            acc += xz[((size_t)b * SEQ_T + tt) * (2 * D_INNER) + d] * cw[d * D_CONV + k];
    }
    float s = acc / (1.f + __expf(-acc)); // SiLU
    xcf[i] = s;
    xch[i] = (_Float16)s;
}

// Extract dt_r slice of x_ssm as f16 for the dt GEMM.
__global__ void extract_dtr_kernel(const float* __restrict__ xssm, _Float16* __restrict__ dtrh) {
    int i = blockIdx.x * blockDim.x + threadIdx.x;
    if (i >= MROWS * DT_RANK) return;
    int m = i / DT_RANK, k = i % DT_RANK;
    dtrh[i] = (_Float16)xssm[(size_t)m * XSSM_W + k];
}

// dt = softplus(dt_lin + bias)
__global__ void softplus_kernel(float* __restrict__ dt, const float* __restrict__ bias) {
    size_t i = (size_t)blockIdx.x * blockDim.x + threadIdx.x;
    if (i >= (size_t)MROWS * D_INNER) return;
    int d = (int)(i % D_INNER);
    float v = dt[i] + bias[d];
    dt[i] = (v > 20.f) ? v : log1pf(__expf(v));
}

// ---------------------------------------------------------------------------
// Selective scan: one wave per (b, d); each lane owns states s=lane, lane+32.
// Recurrence: h = exp(dt*A)*h + dt*x*B ;  y = <h, C> + D*x (shfl_xor reduce).
// ---------------------------------------------------------------------------
__global__ void scan_kernel(const float* __restrict__ dtf, const float* __restrict__ xssm,
                            const float* __restrict__ xcf, const float* __restrict__ state,
                            const float* __restrict__ A_log, const float* __restrict__ Dp,
                            float* __restrict__ yf, float* __restrict__ hf) {
    int wid = blockIdx.x * (blockDim.x >> 5) + (threadIdx.x >> 5);
    int lane = threadIdx.x & 31;
    if (wid >= BATCH * D_INNER) return;
    int b = wid / D_INNER, d = wid % D_INNER;

    float a0 = -__expf(A_log[d * D_STATE + lane]);
    float a1 = -__expf(A_log[d * D_STATE + lane + 32]);
    size_t hbase = ((size_t)b * D_INNER + d) * D_STATE;
    float h0 = state[hbase + lane];
    float h1 = state[hbase + lane + 32];
    float Dv = Dp[d];

    for (int t = 0; t < SEQ_T; ++t) {
        size_t m = (size_t)b * SEQ_T + t;
        float dtv = dtf[m * D_INNER + d];   // broadcast load (same addr all lanes)
        float xv  = xcf[m * D_INNER + d];
        const float* sr = xssm + m * XSSM_W;
        float B0 = sr[DT_RANK + lane];
        float B1 = sr[DT_RANK + 32 + lane];
        float C0 = sr[DT_RANK + D_STATE + lane];
        float C1 = sr[DT_RANK + D_STATE + 32 + lane];
        float dx = dtv * xv;
        h0 = __expf(dtv * a0) * h0 + dx * B0;
        h1 = __expf(dtv * a1) * h1 + dx * B1;
        float part = h0 * C0 + h1 * C1;
#pragma unroll
        for (int mm = 16; mm >= 1; mm >>= 1) part += __shfl_xor(part, mm, 32);
        if (lane == 0) yf[m * D_INNER + d] = part + Dv * xv;
    }
    hf[hbase + lane] = h0;
    hf[hbase + lane + 32] = h1;
}

// y_gated = y * silu(z), emitted as f16 for out_proj. z = xz cols [D_INNER, 2*D_INNER).
__global__ void gate_kernel(const float* __restrict__ yf, const float* __restrict__ xz,
                            _Float16* __restrict__ yh) {
    size_t i = (size_t)blockIdx.x * blockDim.x + threadIdx.x;
    if (i >= (size_t)MROWS * D_INNER) return;
    size_t m = i / D_INNER;
    int d = (int)(i % D_INNER);
    float z = xz[m * (2 * D_INNER) + D_INNER + d];
    float sz = z / (1.f + __expf(-z));
    yh[i] = (_Float16)(yf[i] * sz);
}

// ---------------------------------------------------------------------------
// Host orchestration
// ---------------------------------------------------------------------------
extern "C" void kernel_launch(void* const* d_in, const int* in_sizes, int n_in,
                              void* d_out, int out_size, void* d_ws, size_t ws_size,
                              hipStream_t stream) {
    (void)in_sizes; (void)n_in; (void)out_size; (void)ws_size;

    const float* x         = (const float*)d_in[0];
    const float* state     = (const float*)d_in[1];
    const float* in_proj_w = (const float*)d_in[2];
    const float* conv_w    = (const float*)d_in[3];
    const float* conv_b    = (const float*)d_in[4];
    const float* x_proj_w  = (const float*)d_in[5];
    const float* dt_proj_w = (const float*)d_in[6];
    const float* dt_proj_b = (const float*)d_in[7];
    const float* A_log     = (const float*)d_in[8];
    const float* D_param   = (const float*)d_in[9];
    const float* out_proj_w= (const float*)d_in[10];
    const float* ln_g      = (const float*)d_in[11];
    const float* ln_b      = (const float*)d_in[12];

    float* out     = (float*)d_out;
    float* h_final = out + (size_t)BATCH * SEQ_T * D_MODEL; // second tuple output

    // Bump allocator over d_ws (total ~121 MB).
    char* ws = (char*)d_ws;
    size_t off = 0;
    auto alloc = [&](size_t bytes) -> char* {
        char* p = ws + off;
        off += (bytes + 255) & ~(size_t)255;
        return p;
    };
    _Float16* xnh     = (_Float16*)alloc((size_t)MROWS * D_MODEL * 2);
    _Float16* w_in_h  = (_Float16*)alloc((size_t)2 * D_INNER * D_MODEL * 2);
    float*    xz      = (float*)   alloc((size_t)MROWS * 2 * D_INNER * 4);
    _Float16* xch     = (_Float16*)alloc((size_t)MROWS * D_INNER * 2);
    float*    xcf     = (float*)   alloc((size_t)MROWS * D_INNER * 4);
    _Float16* w_xp_h  = (_Float16*)alloc((size_t)XSSM_W * D_INNER * 2);
    float*    xssm    = (float*)   alloc((size_t)MROWS * XSSM_W * 4);
    _Float16* dtrh    = (_Float16*)alloc((size_t)MROWS * DT_RANK * 2);
    _Float16* w_dt_h  = (_Float16*)alloc((size_t)D_INNER * DT_RANK * 2);
    float*    dtf     = (float*)   alloc((size_t)MROWS * D_INNER * 4);
    float*    yf      = (float*)   alloc((size_t)MROWS * D_INNER * 4);
    _Float16* yh      = (_Float16*)alloc((size_t)MROWS * D_INNER * 2);
    _Float16* w_out_h = (_Float16*)alloc((size_t)D_MODEL * D_INNER * 2);

    const int EB = 256;
    auto eblocks = [](size_t n) { return (int)((n + 255) / 256); };

    // Weight conversions f32 -> f16
    cvt_f32_f16<<<eblocks((size_t)2 * D_INNER * D_MODEL), EB, 0, stream>>>(in_proj_w, w_in_h, 2 * D_INNER * D_MODEL);
    cvt_f32_f16<<<eblocks((size_t)XSSM_W * D_INNER), EB, 0, stream>>>(x_proj_w, w_xp_h, XSSM_W * D_INNER);
    cvt_f32_f16<<<eblocks((size_t)D_INNER * DT_RANK), EB, 0, stream>>>(dt_proj_w, w_dt_h, D_INNER * DT_RANK);
    cvt_f32_f16<<<eblocks((size_t)D_MODEL * D_INNER), EB, 0, stream>>>(out_proj_w, w_out_h, D_MODEL * D_INNER);

    // 1. LayerNorm
    ln_f16_kernel<<<MROWS, 256, 0, stream>>>(x, ln_g, ln_b, xnh);

    // 2. in_proj: [2048,1024] x [4096,1024]^T -> xz (64x32 macro-tiles)
    {
        constexpr int tiles = (MROWS / 64) * ((2 * D_INNER) / 32);
        gemm_f16_wmma<4, 2, MROWS, 2 * D_INNER, D_MODEL, false>
            <<<(tiles + 3) / 4, 128, 0, stream>>>(xnh, w_in_h, nullptr, xz);
    }

    // 3. Depthwise causal conv + SiLU
    conv_silu_kernel<<<eblocks((size_t)MROWS * D_INNER), EB, 0, stream>>>(xz, conv_w, conv_b, xch, xcf);

    // 4. x_proj: [2048,2048] x [192,2048]^T -> x_ssm (16x48 macro-tiles)
    {
        constexpr int tiles = (MROWS / 16) * (XSSM_W / 48);
        gemm_f16_wmma<1, 3, MROWS, XSSM_W, D_INNER, false>
            <<<(tiles + 3) / 4, 128, 0, stream>>>(xch, w_xp_h, nullptr, xssm);
    }

    // 5. dt_proj: [2048,64] x [2048,64]^T -> dt_lin (32x32 macro-tiles)
    extract_dtr_kernel<<<eblocks((size_t)MROWS * DT_RANK), EB, 0, stream>>>(xssm, dtrh);
    {
        constexpr int tiles = (MROWS / 32) * (D_INNER / 32);
        gemm_f16_wmma<2, 2, MROWS, D_INNER, DT_RANK, false>
            <<<(tiles + 3) / 4, 128, 0, stream>>>(dtrh, w_dt_h, nullptr, dtf);
    }
    softplus_kernel<<<eblocks((size_t)MROWS * D_INNER), EB, 0, stream>>>(dtf, dt_proj_b);

    // 6. Selective scan (writes y and h_final)
    scan_kernel<<<(BATCH * D_INNER) / 8, 256, 0, stream>>>(dtf, xssm, xcf, state, A_log, D_param, yf, h_final);

    // 7. Gate with silu(z)
    gate_kernel<<<eblocks((size_t)MROWS * D_INNER), EB, 0, stream>>>(yf, xz, yh);

    // 8. out_proj: [2048,2048] x [1024,2048]^T + resid -> out (64x32 macro-tiles)
    {
        constexpr int tiles = (MROWS / 64) * (D_MODEL / 32);
        gemm_f16_wmma<4, 2, MROWS, D_MODEL, D_INNER, true>
            <<<(tiles + 3) / 4, 128, 0, stream>>>(yh, w_out_h, x, out);
    }
}
